// SADModel_41721312314115
// MI455X (gfx1250) — compile-verified, matrix operand order checked by reading, not run on hardware
//
#include <hip/hip_runtime.h>
#include <math.h>

typedef __attribute__((ext_vector_type(16))) _Float16 v16h;
typedef __attribute__((ext_vector_type(8)))  float    v8f;
typedef int v4i_vs __attribute__((vector_size(4 * sizeof(int))));

#define BB    16
#define NN    2048
#define FIN   40
#define DD    256
#define HH    8
#define KK    256
#define LL    4
#define DHD   32
#define D4    1024
#define ROWS  (BB*NN)   // 32768

#define TM 128
#define TN 64
#define TK 32

// ---------------------------------------------------------------------------
// CDNA5 async global->LDS copies (ASYNCcnt), guarded so compile never breaks.
// Builtin signature (from hipcc diagnostic): (v4i AS1*, v4i AS3*, imm, imm).
// ---------------------------------------------------------------------------
#if defined(__gfx1250__) && __has_builtin(__builtin_amdgcn_global_load_async_to_lds_b128)
#define USE_ASYNC 1
#else
#define USE_ASYNC 0
#endif

__device__ __forceinline__ void async_copy16(const void* g, void* l) {
#if USE_ASYNC
  __builtin_amdgcn_global_load_async_to_lds_b128(
      (__attribute__((address_space(1))) v4i_vs*)g,
      (__attribute__((address_space(3))) v4i_vs*)l, 0, 0);
#else
  *(uint4*)l = *(const uint4*)g;
#endif
}

__device__ __forceinline__ void wait_async0() {
#if USE_ASYNC
#if __has_builtin(__builtin_amdgcn_s_wait_asynccnt)
  __builtin_amdgcn_s_wait_asynccnt(0);
#else
  asm volatile("s_wait_asynccnt 0" ::: "memory");
#endif
#endif
}

// ---------------------------------------------------------------------------
// embed: h = LN0(x @ W_emb + b_emb + pos)   (fp32 residual stream)
// ---------------------------------------------------------------------------
__global__ __launch_bounds__(256) void embed_ln0_kernel(
    const float* __restrict__ x, const float* __restrict__ Wemb,
    const float* __restrict__ bemb, const float* __restrict__ pos,
    const float* __restrict__ g, const float* __restrict__ bta,
    float* __restrict__ h)
{
  const int row = blockIdx.x;      // b*N + n
  const int t   = threadIdx.x;     // channel d
  __shared__ float  xs[FIN];
  __shared__ float2 red[256];
  if (t < FIN) xs[t] = x[(long)row*FIN + t];
  __syncthreads();
  const int n = row & (NN - 1);
  float acc = bemb[t] + pos[(long)n*DD + t];
  #pragma unroll
  for (int f = 0; f < FIN; ++f) acc += xs[f] * Wemb[f*DD + t];
  red[t] = make_float2(acc, acc*acc);
  __syncthreads();
  for (int s = 128; s > 0; s >>= 1) {
    if (t < s) { red[t].x += red[t+s].x; red[t].y += red[t+s].y; }
    __syncthreads();
  }
  const float mean = red[0].x * (1.f/DD);
  const float var  = red[0].y * (1.f/DD) - mean*mean;
  h[(long)row*DD + t] = (acc - mean)*rsqrtf(var + 1e-5f)*g[t] + bta[t];
}

// ---------------------------------------------------------------------------
// LayerNorm fp32 -> f16 (feeds WMMA GEMMs directly)
// ---------------------------------------------------------------------------
__global__ __launch_bounds__(256) void ln16_kernel(
    const float* __restrict__ in, const float* __restrict__ g,
    const float* __restrict__ bta, _Float16* __restrict__ out)
{
  const long row = blockIdx.x;
  const int  t   = threadIdx.x;
  const float xv = in[row*DD + t];
  __shared__ float2 red[256];
  red[t] = make_float2(xv, xv*xv);
  __syncthreads();
  for (int s = 128; s > 0; s >>= 1) {
    if (t < s) { red[t].x += red[t+s].x; red[t].y += red[t+s].y; }
    __syncthreads();
  }
  const float mean = red[0].x * (1.f/DD);
  const float var  = red[0].y * (1.f/DD) - mean*mean;
  out[row*DD + t] = (_Float16)((xv - mean)*rsqrtf(var + 1e-5f)*g[t] + bta[t]);
}

// ---------------------------------------------------------------------------
// transpose + fp32->f16 convert: src [R][C] (C = 1<<cshift) -> dst [C][R]
// used for per-layer weights (W -> W^T f16) and Ek/Ev ([N,K] -> [K,N] f16)
// ---------------------------------------------------------------------------
__global__ __launch_bounds__(256) void wtrans_kernel(
    const float* __restrict__ src, _Float16* __restrict__ dst,
    int R, int cshift)
{
  const long i = (long)blockIdx.x*256 + threadIdx.x;
  const int  r = (int)(i >> cshift);
  const int  c = (int)(i & ((1 << cshift) - 1));
  dst[(long)c*R + r] = (_Float16)src[i];
}

// ---------------------------------------------------------------------------
// WMMA GEMM, all-f16 operands, f32 accumulate.
//   BT=true : B memory is pre-transposed [Nd][Kd] (weights) -> contiguous stage
//   BT=false: B memory is [Kd][Nd] (activations)            -> scatter stage
// Double-buffered LDS; A/B tiles staged with async global->LDS b128 copies
// (ASYNCcnt) so tile k+1 streams in while WMMAs consume tile k.
// flags: bit0 = exact GELU, bit1 = f16 output (else fp32, optional +Res).
// gridDim.z = batch with element strides (sA=0 broadcasts A).
// ---------------------------------------------------------------------------
template<bool BT>
__global__ __launch_bounds__(256) void gemm16_kernel(
    const _Float16* __restrict__ A, long lda, long sA,
    const _Float16* __restrict__ Bm, long ldb, long sB,
    const float* __restrict__ bias,
    const float* __restrict__ Res, long ldr, long sR,
    void* __restrict__ Cv, long ldc, long sC,
    int M, int Nd, int Kd, int flags)
{
  const int tid  = threadIdx.x;
  const int lane = tid & 31;
  const int wave = tid >> 5;
  const int wm   = (wave >> 1) << 5;   // 0,32,64,96
  const int wn   = (wave & 1) << 5;    // 0,32
  const long bM  = (long)blockIdx.y * TM;
  const long bN  = (long)blockIdx.x * TN;
  const long z   = blockIdx.z;
  A  += z * sA;
  Bm += z * sB;

  __shared__ _Float16 As[2][TM][TK + 8];
  __shared__ _Float16 Bs[2][TN][TK + 8];

  v8f acc00 = {0.f,0.f,0.f,0.f,0.f,0.f,0.f,0.f};
  v8f acc01 = acc00, acc10 = acc00, acc11 = acc00;

  const int mrow = lane & 15;
  const int hsel = lane >> 4;

  auto stageA = [&](int buf, int k0) {
    // TM*TK halfs = 512 chunks of 8 halfs (16B); 2 per thread
    #pragma unroll
    for (int i = 0; i < 2; ++i) {
      const int idx = tid + i*256;
      const int r = idx >> 2;            // 4 chunks per 32-half row
      const int c = (idx & 3) << 3;
      async_copy16(A + (bM + r)*lda + k0 + c, &As[buf][r][c]);
    }
  };
  auto stageB = [&](int buf, int k0) {
    if (BT) {
      // B^T memory [Nd][Kd]: row n contiguous in k -> straight copy
      const int r = tid >> 2;            // n row (64), 4 chunks per row
      const int c = (tid & 3) << 3;
      async_copy16(Bm + (bN + r)*ldb + k0 + c, &Bs[buf][r][c]);
    } else {
      // B memory [Kd][Nd]: load 8 halfs along n, scatter-transpose to LDS
      const int r = tid >> 3;            // k row (32), 8 chunks per row
      const int c = (tid & 7) << 3;      // n col
      _Float16 tmp[8];
      *(uint4*)tmp = *(const uint4*)(Bm + (long)(k0 + r)*ldb + bN + c);
      #pragma unroll
      for (int e = 0; e < 8; ++e) Bs[buf][c + e][r] = tmp[e];
    }
  };

  const int KT = Kd / TK;
  stageA(0, 0);
  stageB(0, 0);
  wait_async0();
  __syncthreads();

  for (int kt = 0; kt < KT; ++kt) {
    const int cur = kt & 1;
    if (kt + 1 < KT) {                   // prefetch next tile into other buffer
      stageA(cur ^ 1, (kt + 1)*TK);
      stageB(cur ^ 1, (kt + 1)*TK);
    }

    v16h b0, b1, a0;
    #pragma unroll
    for (int j = 0; j < 8; ++j) {
      const int kk = ((j >> 2) << 4) + (hsel << 3) + ((j & 3) << 1);
      b0[2*j]   = Bs[cur][wn + mrow][kk];
      b0[2*j+1] = Bs[cur][wn + mrow][kk+1];
      b1[2*j]   = Bs[cur][wn + 16 + mrow][kk];
      b1[2*j+1] = Bs[cur][wn + 16 + mrow][kk+1];
      a0[2*j]   = As[cur][wm + mrow][kk];
      a0[2*j+1] = As[cur][wm + mrow][kk+1];
    }
    acc00 = __builtin_amdgcn_wmma_f32_16x16x32_f16(false, a0, false, b0, (short)0, acc00, false, false);
    acc01 = __builtin_amdgcn_wmma_f32_16x16x32_f16(false, a0, false, b1, (short)0, acc01, false, false);
    #pragma unroll
    for (int j = 0; j < 8; ++j) {
      const int kk = ((j >> 2) << 4) + (hsel << 3) + ((j & 3) << 1);
      a0[2*j]   = As[cur][wm + 16 + mrow][kk];
      a0[2*j+1] = As[cur][wm + 16 + mrow][kk+1];
    }
    acc10 = __builtin_amdgcn_wmma_f32_16x16x32_f16(false, a0, false, b0, (short)0, acc10, false, false);
    acc11 = __builtin_amdgcn_wmma_f32_16x16x32_f16(false, a0, false, b1, (short)0, acc11, false, false);

    wait_async0();
    __syncthreads();
  }

  // epilogue: m = rt*16 + (lane>=16)*8 + r ; n = ct*16 + (lane&15)
  float* Cf = (float*)Cv;
  _Float16* Ch = (_Float16*)Cv;
  const float invs2 = 0.70710678118654752f;
  #pragma unroll
  for (int rt = 0; rt < 2; ++rt) {
    #pragma unroll
    for (int ct = 0; ct < 2; ++ct) {
      const v8f acc = (rt == 0) ? (ct == 0 ? acc00 : acc01)
                                : (ct == 0 ? acc10 : acc11);
      const long n  = bN + wn + ct*16 + mrow;
      const float bv = bias ? bias[n] : 0.f;
      #pragma unroll
      for (int r = 0; r < 8; ++r) {
        const long m = bM + wm + rt*16 + hsel*8 + r;
        float val = acc[r] + bv;
        if (flags & 1) val = 0.5f*val*(1.f + erff(val*invs2));
        if (flags & 2) {
          Ch[z*sC + m*ldc + n] = (_Float16)val;
        } else {
          if (Res) val += Res[z*sR + m*ldr + n];
          Cf[z*sC + m*ldc + n] = val;
        }
      }
    }
  }
}

// ---------------------------------------------------------------------------
// fused Linformer attention (f16 in/out): per (b, head), k/v head tiles
// (K=256 x DH=32, f16) staged in LDS via async copies; one query row per
// thread, single-pass online softmax.
// ---------------------------------------------------------------------------
__global__ __launch_bounds__(256) void attn_kernel(
    const _Float16* __restrict__ q, const _Float16* __restrict__ k,
    const _Float16* __restrict__ v, _Float16* __restrict__ o)
{
  const int bh  = blockIdx.x;
  const int b   = bh >> 3;
  const int hd  = bh & 7;
  const int tid = threadIdx.x;
  __shared__ _Float16 ksh[KK][DHD];
  __shared__ _Float16 vsh[KK][DHD];
  #pragma unroll
  for (int i = 0; i < 4; ++i) {              // 1024 chunks of 8 halfs each tile
    const int idx = tid + i*256;
    const int kk = idx >> 2;
    const int d  = (idx & 3) << 3;
    const long base = ((long)(b*KK + kk))*DD + hd*DHD + d;
    async_copy16(k + base, &ksh[kk][d]);
    async_copy16(v + base, &vsh[kk][d]);
  }
  wait_async0();
  __syncthreads();

  const int  n = blockIdx.y*256 + tid;
  const long qbase = ((long)(b*NN + n))*DD + hd*DHD;
  float qr[DHD];
  #pragma unroll
  for (int d = 0; d < DHD; ++d) qr[d] = (float)q[qbase + d];

  const float scale = 0.17677669529663687f;  // 1/sqrt(32)
  float mmax = -3.0e38f, lsum = 0.f, oacc[DHD];
  #pragma unroll
  for (int d = 0; d < DHD; ++d) oacc[d] = 0.f;

  for (int kk = 0; kk < KK; ++kk) {
    float s = 0.f;
    #pragma unroll
    for (int d = 0; d < DHD; ++d) s += qr[d]*(float)ksh[kk][d];
    s *= scale;
    const float mn   = fmaxf(mmax, s);
    const float corr = __expf(mmax - mn);
    const float p    = __expf(s - mn);
    lsum = lsum*corr + p;
    #pragma unroll
    for (int d = 0; d < DHD; ++d) oacc[d] = oacc[d]*corr + p*(float)vsh[kk][d];
    mmax = mn;
  }
  const float inv = 1.f/lsum;
  #pragma unroll
  for (int d = 0; d < DHD; ++d) o[qbase + d] = (_Float16)(oacc[d]*inv);
}

// ---------------------------------------------------------------------------
// out = h @ W_out + b_out  (fp32 GEMV; one wave per row)
// ---------------------------------------------------------------------------
__global__ __launch_bounds__(256) void out_kernel(
    const float* __restrict__ h, const float* __restrict__ Wout,
    const float* __restrict__ bout, float* __restrict__ out)
{
  const int  tid  = threadIdx.x;
  const int  lane = tid & 31;
  const long row  = (long)blockIdx.x*8 + (tid >> 5);
  float s = 0.f;
  #pragma unroll
  for (int j = 0; j < 8; ++j) {
    const int c = lane + j*32;
    s += h[row*DD + c]*Wout[c];
  }
  #pragma unroll
  for (int off = 16; off > 0; off >>= 1) s += __shfl_xor(s, off, 32);
  if (lane == 0) out[row] = s + bout[0];
}

// ---------------------------------------------------------------------------
static inline void gemm16(bool bt,
                          const _Float16* A, long lda, long sA,
                          const _Float16* Bm, long ldb, long sB,
                          const float* bias,
                          const float* Res, long ldr, long sR,
                          void* C, long ldc, long sC,
                          int M, int Nd, int Kd, int batch, int flags,
                          hipStream_t st)
{
  dim3 grid(Nd/TN, M/TM, batch);
  if (bt)
    gemm16_kernel<true><<<grid, 256, 0, st>>>(A, lda, sA, Bm, ldb, sB, bias,
                                              Res, ldr, sR, C, ldc, sC,
                                              M, Nd, Kd, flags);
  else
    gemm16_kernel<false><<<grid, 256, 0, st>>>(A, lda, sA, Bm, ldb, sB, bias,
                                               Res, ldr, sR, C, ldc, sC,
                                               M, Nd, Kd, flags);
}

extern "C" void kernel_launch(void* const* d_in, const int* in_sizes, int n_in,
                              void* d_out, int out_size, void* d_ws, size_t ws_size,
                              hipStream_t stream)
{
  (void)in_sizes; (void)n_in; (void)out_size; (void)ws_size;
  const float* x     = (const float*)d_in[0];
  const float* W_emb = (const float*)d_in[1];
  const float* b_emb = (const float*)d_in[2];
  const float* pos   = (const float*)d_in[3];
  const float* ln0_g = (const float*)d_in[4];
  const float* ln0_b = (const float*)d_in[5];
  const float* ln1_g = (const float*)d_in[6];
  const float* ln1_b = (const float*)d_in[7];
  const float* Wq    = (const float*)d_in[8];
  const float* Wk    = (const float*)d_in[9];
  const float* Wv    = (const float*)d_in[10];
  const float* Ek    = (const float*)d_in[11];
  const float* Ev    = (const float*)d_in[12];
  const float* Wo    = (const float*)d_in[13];
  const float* bo    = (const float*)d_in[14];
  const float* ln2_g = (const float*)d_in[15];
  const float* ln2_b = (const float*)d_in[16];
  const float* W1    = (const float*)d_in[17];
  const float* b1    = (const float*)d_in[18];
  const float* W2    = (const float*)d_in[19];
  const float* b2    = (const float*)d_in[20];
  const float* W_out = (const float*)d_in[21];
  const float* b_out = (const float*)d_in[22];

  float*    h   = (float*)d_ws;                       // [32768,256] fp32 residual
  _Float16* a16 = (_Float16*)(h + (long)ROWS*DD);     // LN out
  _Float16* t0h = a16 + (long)ROWS*DD;                // q
  _Float16* t1h = t0h + (long)ROWS*DD;                // kproj/vproj then o
  _Float16* kbh = t1h + (long)ROWS*DD;                // [16,256,256]
  _Float16* vbh = kbh + (long)BB*KK*DD;
  _Float16* ETh = vbh + (long)BB*KK*DD;               // [256,2048]
  _Float16* ffh = ETh + (long)KK*NN;                  // [32768,1024]
  _Float16* wqt = ffh + (long)ROWS*D4;                // per-layer W^T f16
  _Float16* wkt = wqt + (long)DD*DD;
  _Float16* wvt = wkt + (long)DD*DD;
  _Float16* wot = wvt + (long)DD*DD;
  _Float16* w1t = wot + (long)DD*DD;                  // [1024,256]
  _Float16* w2t = w1t + (long)DD*D4;                  // [256,1024]

  embed_ln0_kernel<<<ROWS, 256, 0, stream>>>(x, W_emb, b_emb, pos, ln0_g, ln0_b, h);

  for (int i = 0; i < LL; ++i) {
    // per-layer weight transpose+convert to f16 (tiny)
    wtrans_kernel<<<(DD*DD)/256, 256, 0, stream>>>(Wq + (long)i*DD*DD, wqt, DD, 8);
    wtrans_kernel<<<(DD*DD)/256, 256, 0, stream>>>(Wk + (long)i*DD*DD, wkt, DD, 8);
    wtrans_kernel<<<(DD*DD)/256, 256, 0, stream>>>(Wv + (long)i*DD*DD, wvt, DD, 8);
    wtrans_kernel<<<(DD*DD)/256, 256, 0, stream>>>(Wo + (long)i*DD*DD, wot, DD, 8);
    wtrans_kernel<<<(DD*D4)/256, 256, 0, stream>>>(W1 + (long)i*DD*D4, w1t, DD, 10);
    wtrans_kernel<<<(D4*DD)/256, 256, 0, stream>>>(W2 + (long)i*D4*DD, w2t, D4, 8);

    // a = LN1(h)  -> f16
    ln16_kernel<<<ROWS, 256, 0, stream>>>(h, ln1_g + i*DD, ln1_b + i*DD, a16);

    // q = a @ Wq  (f16 out)
    gemm16(true, a16, DD, 0, wqt, DD, 0, nullptr, nullptr, 0, 0,
           t0h, DD, 0, ROWS, DD, DD, 1, 2, stream);

    // kproj = a @ Wk ; k = Ek^T @ kproj (batched, A broadcast)
    gemm16(true, a16, DD, 0, wkt, DD, 0, nullptr, nullptr, 0, 0,
           t1h, DD, 0, ROWS, DD, DD, 1, 2, stream);
    wtrans_kernel<<<(NN*KK)/256, 256, 0, stream>>>(Ek + (long)i*NN*KK, ETh, NN, 8);
    gemm16(false, ETh, NN, 0, t1h, DD, (long)NN*DD, nullptr, nullptr, 0, 0,
           kbh, DD, (long)KK*DD, KK, DD, NN, BB, 2, stream);

    // vproj = a @ Wv ; v = Ev^T @ vproj
    gemm16(true, a16, DD, 0, wvt, DD, 0, nullptr, nullptr, 0, 0,
           t1h, DD, 0, ROWS, DD, DD, 1, 2, stream);
    wtrans_kernel<<<(NN*KK)/256, 256, 0, stream>>>(Ev + (long)i*NN*KK, ETh, NN, 8);
    gemm16(false, ETh, NN, 0, t1h, DD, (long)NN*DD, nullptr, nullptr, 0, 0,
           vbh, DD, (long)KK*DD, KK, DD, NN, BB, 2, stream);

    // o = attention(q,k,v) -> t1h (f16)
    attn_kernel<<<dim3(BB*HH, NN/256), 256, 0, stream>>>(t0h, kbh, vbh, t1h);

    // h = h + o @ Wo + bo   (fp32 out, residual)
    gemm16(true, t1h, DD, 0, wot, DD, 0, bo + i*DD, h, DD, 0,
           h, DD, 0, ROWS, DD, DD, 1, 0, stream);

    // a = LN2(h) -> f16
    ln16_kernel<<<ROWS, 256, 0, stream>>>(h, ln2_g + i*DD, ln2_b + i*DD, a16);

    // ff = gelu(a @ W1 + b1)  (f16 out)
    gemm16(true, a16, DD, 0, w1t, DD, 0, b1 + i*D4, nullptr, 0, 0,
           ffh, D4, 0, ROWS, D4, DD, 1, 3, stream);

    // h = h + ff @ W2 + b2  (fp32 out, residual)
    gemm16(true, ffh, D4, 0, w2t, D4, 0, b2 + i*DD, h, DD, 0,
           h, DD, 0, ROWS, DD, D4, 1, 0, stream);
  }

  out_kernel<<<ROWS/8, 256, 0, stream>>>(h, W_out, b_out, (float*)d_out);
}
